// TFCoaT_33217277067354
// MI455X (gfx1250) — matrix operand order repaired and checked by reading, not compiled, not run on hardware
//
#include <hip/hip_runtime.h>
#include <hip/hip_bf16.h>
#include <cstddef>

// ---------------------------------------------------------------------------
// Problem constants (from reference)
// ---------------------------------------------------------------------------
#define Bc 32
#define Nn 1025
#define Cc 512
#define Hh 8
#define HD 64
#define MTOT (Bc * Nn)          // 32800 = 32 * 1025, multiple of 32
#define SCALEF 0.125f           // 64^-0.5

typedef __attribute__((ext_vector_type(16))) __bf16 v16bf;
typedef __attribute__((ext_vector_type(8)))  __bf16 v8bf;
typedef __attribute__((ext_vector_type(8)))  float  v8f;

__device__ __forceinline__ __bf16 f2bf(float f) { return (__bf16)f; }

// ---------------------------------------------------------------------------
// Converters
// ---------------------------------------------------------------------------
__global__ void cvt_f32_bf16(const float* __restrict__ in, __bf16* __restrict__ out, size_t n) {
    size_t i = (size_t)blockIdx.x * 256 + threadIdx.x;
    if (i < n) out[i] = f2bf(in[i]);
}

// in: rows x cols (row-major f32)  ->  out: cols x rows (row-major bf16)
__global__ void cvt_transpose_bf16(const float* __restrict__ in, __bf16* __restrict__ out,
                                   int rows, int cols) {
    size_t i = (size_t)blockIdx.x * 256 + threadIdx.x;
    size_t total = (size_t)rows * cols;
    if (i < total) {
        int c = (int)(i % cols);
        int r = (int)(i / cols);
        out[(size_t)c * rows + r] = f2bf(in[i]);
    }
}

// ---------------------------------------------------------------------------
// Fragment loader helpers (ISA 7.12.2 layouts, wave32)
// A 16x32 bf16: lane half h = lane>>4, row M = lane&15
//   elems 0..7  : K = k0 + 8h + e        elems 8..15 : K = k0 + 16 + 8h + (e-8)
// B 32x16 bf16: col N = lane&15, same K order as A
//   elems 0..7  : K = k0 + 8h + e        elems 8..15 : K = k0 + 16 + 8h + (e-8)
// ---------------------------------------------------------------------------
__device__ __forceinline__ v16bf load_a_frag(const __bf16* __restrict__ Arow, int k0, int half) {
    v8bf lo = *(const v8bf*)(Arow + k0 + 8 * half);
    v8bf hi = *(const v8bf*)(Arow + k0 + 16 + 8 * half);
    v16bf a;
#pragma unroll
    for (int i = 0; i < 8; ++i) { a[i] = lo[i]; a[i + 8] = hi[i]; }
    return a;
}

__device__ __forceinline__ v16bf load_b_frag(const __bf16* __restrict__ Brow, int k0, int half) {
    v8bf lo = *(const v8bf*)(Brow + k0 + 8 * half);
    v8bf hi = *(const v8bf*)(Brow + k0 + 16 + 8 * half);
    v16bf b;
#pragma unroll
    for (int i = 0; i < 8; ++i) { b[i] = lo[i]; b[i + 8] = hi[i]; }
    return b;
}

// ---------------------------------------------------------------------------
// WMMA GEMM:  Out[M x Nc](f32) = A[M x K](bf16) @ Bt[Nc x K](bf16)^T + bias
// 2x2 register tile per wave (32x32 output), 8 waves/block along N
//   -> block tile 32M x 256N, 4 v_wmma per 8 b128 fragment loads.
// C/D layout: elem r -> M = r + 8*(lane>>4), N = lane&15
// ---------------------------------------------------------------------------
__global__ __launch_bounds__(256) void wmma_gemm_bf16(
    const __bf16* __restrict__ A, const __bf16* __restrict__ Bt,
    const float* __restrict__ bias, float* __restrict__ Out, int K, int Nc) {

    const int lane   = threadIdx.x & 31;
    const int waveId = threadIdx.x >> 5;
    const int mBase  = blockIdx.y * 32;
    const int nBase  = blockIdx.x * 256 + waveId * 32;
    const int half   = lane >> 4;
    const int l15    = lane & 15;

    const __bf16* Arow0 = A  + (size_t)(mBase + l15) * K;
    const __bf16* Arow1 = A  + (size_t)(mBase + 16 + l15) * K;
    const __bf16* Brow0 = Bt + (size_t)(nBase + l15) * K;
    const __bf16* Brow1 = Bt + (size_t)(nBase + 16 + l15) * K;

    v8f acc00 = {}, acc01 = {}, acc10 = {}, acc11 = {};

#pragma unroll 2
    for (int k0 = 0; k0 < K; k0 += 32) {
        // speculative prefetch two k-steps ahead (global_prefetch_b8, no counters)
        __builtin_prefetch(Arow0 + k0 + 64, 0, 1);
        __builtin_prefetch(Arow1 + k0 + 64, 0, 1);
        __builtin_prefetch(Brow0 + k0 + 64, 0, 1);
        __builtin_prefetch(Brow1 + k0 + 64, 0, 1);

        v16bf a0 = load_a_frag(Arow0, k0, half);
        v16bf a1 = load_a_frag(Arow1, k0, half);
        v16bf b0 = load_b_frag(Brow0, k0, half);
        v16bf b1 = load_b_frag(Brow1, k0, half);

        acc00 = __builtin_amdgcn_wmma_f32_16x16x32_bf16(false, a0, false, b0, (short)0, acc00, false, false);
        acc01 = __builtin_amdgcn_wmma_f32_16x16x32_bf16(false, a0, false, b1, (short)0, acc01, false, false);
        acc10 = __builtin_amdgcn_wmma_f32_16x16x32_bf16(false, a1, false, b0, (short)0, acc10, false, false);
        acc11 = __builtin_amdgcn_wmma_f32_16x16x32_bf16(false, a1, false, b1, (short)0, acc11, false, false);
    }

    const int col0 = nBase + l15;
    const int col1 = nBase + 16 + l15;
    const float bv0 = bias[col0];
    const float bv1 = bias[col1];
#pragma unroll
    for (int r = 0; r < 8; ++r) {
        const int row0 = mBase + r + 8 * half;
        const int row1 = mBase + 16 + r + 8 * half;
        Out[(size_t)row0 * Nc + col0] = acc00[r] + bv0;
        Out[(size_t)row0 * Nc + col1] = acc01[r] + bv1;
        Out[(size_t)row1 * Nc + col0] = acc10[r] + bv0;
        Out[(size_t)row1 * Nc + col1] = acc11[r] + bv1;
    }
}

// ---------------------------------------------------------------------------
// Softmax stats over N for each (b,h,d): column max and 1/sum(exp)
// qkv layout: (b, n, s, h, d) flat, s: 0=q 1=k 2=v, row stride 1536
// ---------------------------------------------------------------------------
__global__ __launch_bounds__(256) void kstat_kernel(const float* __restrict__ qkv,
                                                    float* __restrict__ kmax,
                                                    float* __restrict__ kinv) {
    const int bh = blockIdx.x;          // b*8 + h
    const int b = bh >> 3, h = bh & 7;
    const int t = threadIdx.x;
    const int d = t & 63, chunk = t >> 6;  // 4 chunks over N

    __shared__ float red[256];
    __shared__ float mx_s[64];

    const float* kb = qkv + (size_t)b * Nn * 1536 + 512 + h * 64 + d;

    float m = -3.4e38f;
    for (int n = chunk; n < Nn; n += 4) m = fmaxf(m, kb[(size_t)n * 1536]);
    red[t] = m;
    __syncthreads();
    if (chunk == 0)
        mx_s[d] = fmaxf(fmaxf(red[d], red[d + 64]), fmaxf(red[d + 128], red[d + 192]));
    __syncthreads();

    const float mval = mx_s[d];
    float s = 0.f;
    for (int n = chunk; n < Nn; n += 4) s += __expf(kb[(size_t)n * 1536] - mval);
    red[t] = s;
    __syncthreads();
    if (chunk == 0) {
        s = red[d] + red[d + 64] + red[d + 128] + red[d + 192];
        kmax[bh * 64 + d] = mval;
        kinv[bh * 64 + d] = 1.0f / s;
    }
}

// ---------------------------------------------------------------------------
// ksv[b,h,kd,vd] = sum_n softmax_k[b,n,h,kd] * v[b,n,h,vd]
// One block per (b,h); thread t owns (kd = t>>2, vd = (t&3)*16 .. +15)
// ---------------------------------------------------------------------------
__global__ __launch_bounds__(256) void ksv_kernel(const float* __restrict__ qkv,
                                                  const float* __restrict__ kmax,
                                                  const float* __restrict__ kinv,
                                                  float* __restrict__ ksv) {
    const int bh = blockIdx.x;
    const int b = bh >> 3, h = bh & 7;
    const int t = threadIdx.x;

    __shared__ float ks[4][64];
    __shared__ float vs[4][64];
    __shared__ float mx[64], inv[64];
    if (t < 64) { mx[t] = kmax[bh * 64 + t]; inv[t] = kinv[bh * 64 + t]; }
    __syncthreads();

    const int jload = t >> 6, dload = t & 63;
    const float* base = qkv + (size_t)b * Nn * 1536 + h * 64;

    float acc[16];
#pragma unroll
    for (int i = 0; i < 16; ++i) acc[i] = 0.f;
    const int kd = t >> 2, vd0 = (t & 3) * 16;

    for (int n0 = 0; n0 < Nn; n0 += 4) {
        const int n = n0 + jload;
        if (n < Nn) {
            const float* row = base + (size_t)n * 1536;
            ks[jload][dload] = __expf(row[512 + dload] - mx[dload]) * inv[dload];
            vs[jload][dload] = row[1024 + dload];
        } else {
            ks[jload][dload] = 0.f;
            vs[jload][dload] = 0.f;
        }
        __syncthreads();
#pragma unroll
        for (int j = 0; j < 4; ++j) {
            const float kv = ks[j][kd];
#pragma unroll
            for (int c = 0; c < 16; ++c) acc[c] += kv * vs[j][vd0 + c];
        }
        __syncthreads();
    }
    float* outp = ksv + ((size_t)bh * 64 + kd) * 64 + vd0;
#pragma unroll
    for (int c = 0; c < 16; ++c) outp[c] = acc[c];
}

// ---------------------------------------------------------------------------
// Depthwise convs on v image (B,32,32,512), splits 128/192/192 with 3/5/7 kern
// One block per (b, y, x); 256 threads, 2 channels each.
// ---------------------------------------------------------------------------
__global__ __launch_bounds__(256) void dwconv_kernel(
    const float* __restrict__ qkv,
    const float* __restrict__ ck3, const float* __restrict__ cb3,
    const float* __restrict__ ck5, const float* __restrict__ cb5,
    const float* __restrict__ ck7, const float* __restrict__ cb7,
    float* __restrict__ convv) {

    const int p = blockIdx.x;           // b*1024 + y*32 + x
    const int b = p >> 10, yx = p & 1023;
    const int y = yx >> 5, x = yx & 31;

    for (int c = threadIdx.x; c < 512; c += 256) {
        int rad, cloc, cstride;
        const float* kw;
        float acc;
        if (c < 128)      { rad = 1; kw = ck3; cloc = c;       cstride = 128; acc = cb3[cloc]; }
        else if (c < 320) { rad = 2; kw = ck5; cloc = c - 128; cstride = 192; acc = cb5[cloc]; }
        else              { rad = 3; kw = ck7; cloc = c - 320; cstride = 192; acc = cb7[cloc]; }
        const int kdim = 2 * rad + 1;
        for (int dy = -rad; dy <= rad; ++dy) {
            const int yy = y + dy;
            if (yy < 0 || yy >= 32) continue;
            for (int dx = -rad; dx <= rad; ++dx) {
                const int xx = x + dx;
                if (xx < 0 || xx >= 32) continue;
                const float w = kw[((dy + rad) * kdim + (dx + rad)) * cstride + cloc];
                const float vv = qkv[(((size_t)b * Nn + 1 + yy * 32 + xx) * 3 + 2) * 512 + c];
                acc += w * vv;
            }
        }
        convv[((size_t)b * 1024 + yx) * 512 + c] = acc;
    }
}

// ---------------------------------------------------------------------------
// att = SCALE * (q @ ksv) + crpe, written as bf16 (input of proj GEMM)
// One block per (b,n); 128 threads, 4 channels each.
// ---------------------------------------------------------------------------
__global__ __launch_bounds__(128) void att_kernel(const float* __restrict__ qkv,
                                                  const float* __restrict__ ksv,
                                                  const float* __restrict__ convv,
                                                  __bf16* __restrict__ attb) {
    const int bn = blockIdx.x;
    const int b = bn / Nn, n = bn % Nn;

    __shared__ float qrow[512];
    const float* qp = qkv + (size_t)bn * 1536;   // s = 0 (q)
    for (int i = threadIdx.x; i < 512; i += 128) qrow[i] = qp[i];
    __syncthreads();

#pragma unroll
    for (int j = 0; j < 4; ++j) {
        const int c = threadIdx.x + 128 * j;
        const int h = c >> 6, d = c & 63;
        const float* kp = ksv + ((size_t)(b * 8 + h) * 64) * 64 + d;
        float f = 0.f;
#pragma unroll 8
        for (int kd = 0; kd < 64; ++kd) f += qrow[h * 64 + kd] * kp[(size_t)kd * 64];
        float crpe = 0.f;
        if (n > 0) crpe = qrow[c] * convv[((size_t)b * 1024 + (n - 1)) * 512 + c];
        attb[(size_t)bn * 512 + c] = f2bf(SCALEF * f + crpe);
    }
}

// ---------------------------------------------------------------------------
// Launch
// ---------------------------------------------------------------------------
extern "C" void kernel_launch(void* const* d_in, const int* in_sizes, int n_in,
                              void* d_out, int out_size, void* d_ws, size_t ws_size,
                              hipStream_t stream) {
    const float* x     = (const float*)d_in[0];
    const float* Wqkv  = (const float*)d_in[1];
    const float* bqkv  = (const float*)d_in[2];
    const float* Wproj = (const float*)d_in[3];
    const float* bproj = (const float*)d_in[4];
    const float* ck3   = (const float*)d_in[5];
    const float* cb3   = (const float*)d_in[6];
    const float* ck5   = (const float*)d_in[7];
    const float* cb5   = (const float*)d_in[8];
    const float* ck7   = (const float*)d_in[9];
    const float* cb7   = (const float*)d_in[10];
    float* out = (float*)d_out;

    // Workspace carve-up (256B aligned)
    char* ws = (char*)d_ws;
    size_t off = 0;
    auto alloc = [&](size_t bytes) {
        char* p = ws + off;
        off += (bytes + 255) & ~(size_t)255;
        return p;
    };
    __bf16* xb      = (__bf16*)alloc((size_t)MTOT * Cc * 2);        // 33.6 MB
    __bf16* wqkvT   = (__bf16*)alloc((size_t)3 * Cc * Cc * 2);      // 1.6 MB (1536 x 512)
    __bf16* wprojT  = (__bf16*)alloc((size_t)Cc * Cc * 2);          // 0.5 MB (512 x 512)
    float*  qkv     = (float*) alloc((size_t)MTOT * 3 * Cc * 4);    // 201.5 MB
    float*  kmax    = (float*) alloc((size_t)Bc * Hh * HD * 4);
    float*  kinv    = (float*) alloc((size_t)Bc * Hh * HD * 4);
    float*  ksv     = (float*) alloc((size_t)Bc * Hh * HD * HD * 4);// 4 MB
    float*  convv   = (float*) alloc((size_t)Bc * 1024 * Cc * 4);   // 67 MB
    __bf16* attb    = (__bf16*)alloc((size_t)MTOT * Cc * 2);        // 33.6 MB
    (void)ws_size; (void)n_in; (void)in_sizes; (void)out_size;

    // 1) converts
    {
        size_t nx = (size_t)MTOT * Cc;
        cvt_f32_bf16<<<(unsigned)((nx + 255) / 256), 256, 0, stream>>>(x, xb, nx);
        size_t nw = (size_t)Cc * 3 * Cc;
        cvt_transpose_bf16<<<(unsigned)((nw + 255) / 256), 256, 0, stream>>>(Wqkv, wqkvT, Cc, 3 * Cc);
        size_t np = (size_t)Cc * Cc;
        cvt_transpose_bf16<<<(unsigned)((np + 255) / 256), 256, 0, stream>>>(Wproj, wprojT, Cc, Cc);
    }

    // 2) QKV GEMM: (32800 x 512) @ (512 x 1536) + bqkv -> qkv (f32)
    {
        dim3 grid(3 * Cc / 256, MTOT / 32);   // (6, 1025)
        wmma_gemm_bf16<<<grid, 256, 0, stream>>>(xb, wqkvT, bqkv, qkv, Cc, 3 * Cc);
    }

    // 3) softmax stats over N per (b,h,d)
    kstat_kernel<<<Bc * Hh, 256, 0, stream>>>(qkv, kmax, kinv);

    // 4) ksv = softmax(k)^T @ v per (b,h)
    ksv_kernel<<<Bc * Hh, 256, 0, stream>>>(qkv, kmax, kinv, ksv);

    // 5) depthwise convs on v image
    dwconv_kernel<<<Bc * 1024, 256, 0, stream>>>(qkv, ck3, cb3, ck5, cb5, ck7, cb7, convv);

    // 6) att = SCALE*(q@ksv) + crpe  -> bf16
    att_kernel<<<Bc * Nn, 128, 0, stream>>>(qkv, ksv, convv, attb);

    // 7) proj GEMM: (32800 x 512) @ (512 x 512) + bproj -> out (f32)
    {
        dim3 grid(Cc / 256, MTOT / 32);       // (2, 1025)
        wmma_gemm_bf16<<<grid, 256, 0, stream>>>(attb, wprojT, bproj, out, Cc, Cc);
    }
}
